// AttentionVisual_66769561584262
// MI455X (gfx1250) — matrix operand order repaired and checked by reading, not compiled, not run on hardware
//
#include <hip/hip_runtime.h>
#include <hip/hip_bf16.h>

// ---------------------------------------------------------------------------
// Problem dims (fixed by reference)
#define VIS 1024   // vis encoder dim (K of big GEMM)
#define HID 512    // sentence hidden dim
#define ATT 512    // attention dim (N of big GEMM)
#define NB  64     // batch
#define NL  1024   // sequence length

typedef __attribute__((ext_vector_type(16))) __bf16 v16bf;
typedef __attribute__((ext_vector_type(8)))  __bf16 v8bf;
typedef __attribute__((ext_vector_type(8)))  float  v8f;
typedef __attribute__((ext_vector_type(4)))  float  v4f;

// LDS A-panel row stride (bf16 elements). 1024 + 8 pad: row stride = 516
// dwords, 516 % 64 = 4 -> 16 rows * 4 banks cover all 64 banks, so the
// per-row ds_load_b128 pattern is bank-conflict-free.
#define A_STRIDE 1032
#define MROWS    32     // l-rows per block (2 WMMA M-tiles)

// ---------------------------------------------------------------------------
// Kernel 0a: W_enc [VIS][ATT] f32  ->  Wt [ATT][VIS] bf16 (transposed)
__global__ __launch_bounds__(256) void k_prep_wt(const float* __restrict__ W,
                                                 __bf16* __restrict__ Wt) {
  int idx = blockIdx.x * 256 + threadIdx.x;   // over ATT*VIS
  int a = idx >> 10;                          // 0..511
  int v = idx & 1023;                         // 0..1023
  Wt[idx] = (__bf16)W[v * ATT + a];
}

// Kernel 0b: dec_out[b][a] = sum_h H[b][h] * W_dec[h][a] + b_dec[a]
__global__ __launch_bounds__(256) void k_dec_proj(const float* __restrict__ H,
                                                  const float* __restrict__ Wd,
                                                  const float* __restrict__ bd,
                                                  float* __restrict__ dec_out) {
  int idx = blockIdx.x * 256 + threadIdx.x;   // over NB*ATT
  int b = idx >> 9;
  int a = idx & 511;
  float s = bd[a];
  const float* h = H + b * HID;
#pragma unroll 4
  for (int k = 0; k < HID; ++k) s += h[k] * Wd[k * ATT + a];
  dec_out[idx] = s;
}

// ---------------------------------------------------------------------------
// Kernel 1: fused  logits[b][l] = w_full . tanh(vis[b][l] @ W_enc + b_enc
//                                               + dec_out[b]) + b_full
// Block: 256 threads (8 waves). Each block: one b, 32 rows of l (2 M-tiles).
// Each wave owns 4 N-tiles of 16 (8 waves * 64 = 512 = ATT) for both M-tiles.
__global__ __launch_bounds__(256) void k_att_gemm(
    const float*  __restrict__ vis,      // [NB][NL][VIS]
    const __bf16* __restrict__ Wt,       // [ATT][VIS] bf16
    const float*  __restrict__ b_enc,    // [ATT]
    const float*  __restrict__ dec_out,  // [NB][ATT]
    const float*  __restrict__ w_full,   // [ATT]
    const float*  __restrict__ b_full,   // [1]
    float*        __restrict__ logits)   // [NB][NL]
{
  __shared__ __bf16 As[MROWS * A_STRIDE];  // 32 x 1024 bf16 A-panel (padded)
  __shared__ float  s_logit[MROWS];

  const int tid  = threadIdx.x;
  const int lane = tid & 31;
  const int wave = tid >> 5;
  const int b    = blockIdx.y;
  const int l0   = blockIdx.x * MROWS;

  if (tid < MROWS) s_logit[tid] = 0.0f;

  // ---- stage A panel: 32 rows x VIS f32 -> bf16 in LDS ----
  {
    const int row = tid >> 3;            // 0..31
    const int kb  = (tid & 7) * 128;     // 0..896
    const float* src = vis + ((size_t)b * NL + (l0 + row)) * VIS + kb;
    __bf16* dst = As + row * A_STRIDE + kb;
#pragma unroll
    for (int i = 0; i < 128; i += 8) {
      v4f f0 = *(const v4f*)(src + i);
      v4f f1 = *(const v4f*)(src + i + 4);
      v8bf h;
      h[0] = (__bf16)f0[0]; h[1] = (__bf16)f0[1];
      h[2] = (__bf16)f0[2]; h[3] = (__bf16)f0[3];
      h[4] = (__bf16)f1[0]; h[5] = (__bf16)f1[1];
      h[6] = (__bf16)f1[2]; h[7] = (__bf16)f1[3];
      *(v8bf*)(dst + i) = h;
    }
  }
  __syncthreads();

  // ---- WMMA main loop ----
  const int m  = lane & 15;   // row within tile (A) / col within tile (B,C)
  const int hi = lane >> 4;   // half-wave select

  v8f acc[2][4] = {};         // [M-tile][N-tile] 16x16 f32 accumulators

  // B: lane wants column (wave*64 + t*16 + m), 16 consecutive K starting at
  // kc + hi*16  -> contiguous 32B in Wt (32B aligned).
  const __bf16* bbase  = Wt + (size_t)(wave * 64 + m) * VIS + hi * 16;
  // A: lane holds row (mt*16 + m); 16 consecutive K starting at kc + hi*16.
  const __bf16* abase0 = As + m * A_STRIDE + hi * 16;
  const __bf16* abase1 = abase0 + 16 * A_STRIDE;

  for (int kc = 0; kc < VIS; kc += 32) {
    v8bf a0lo = *(const v8bf*)(abase0 + kc);
    v8bf a0hi = *(const v8bf*)(abase0 + kc + 8);
    v8bf a1lo = *(const v8bf*)(abase1 + kc);
    v8bf a1hi = *(const v8bf*)(abase1 + kc + 8);
    v16bf a0 = __builtin_shufflevector(a0lo, a0hi,
        0, 1, 2, 3, 4, 5, 6, 7, 8, 9, 10, 11, 12, 13, 14, 15);
    v16bf a1 = __builtin_shufflevector(a1lo, a1hi,
        0, 1, 2, 3, 4, 5, 6, 7, 8, 9, 10, 11, 12, 13, 14, 15);

    v16bf b0 = *(const v16bf*)(bbase + kc);
    v16bf b1 = *(const v16bf*)(bbase + (size_t)16 * VIS + kc);
    v16bf b2 = *(const v16bf*)(bbase + (size_t)32 * VIS + kc);
    v16bf b3 = *(const v16bf*)(bbase + (size_t)48 * VIS + kc);

    acc[0][0] = __builtin_amdgcn_wmma_f32_16x16x32_bf16(false, a0, false, b0,
                                                        (short)0, acc[0][0], false, false);
    acc[0][1] = __builtin_amdgcn_wmma_f32_16x16x32_bf16(false, a0, false, b1,
                                                        (short)0, acc[0][1], false, false);
    acc[0][2] = __builtin_amdgcn_wmma_f32_16x16x32_bf16(false, a0, false, b2,
                                                        (short)0, acc[0][2], false, false);
    acc[0][3] = __builtin_amdgcn_wmma_f32_16x16x32_bf16(false, a0, false, b3,
                                                        (short)0, acc[0][3], false, false);
    acc[1][0] = __builtin_amdgcn_wmma_f32_16x16x32_bf16(false, a1, false, b0,
                                                        (short)0, acc[1][0], false, false);
    acc[1][1] = __builtin_amdgcn_wmma_f32_16x16x32_bf16(false, a1, false, b1,
                                                        (short)0, acc[1][1], false, false);
    acc[1][2] = __builtin_amdgcn_wmma_f32_16x16x32_bf16(false, a1, false, b2,
                                                        (short)0, acc[1][2], false, false);
    acc[1][3] = __builtin_amdgcn_wmma_f32_16x16x32_bf16(false, a1, false, b3,
                                                        (short)0, acc[1][3], false, false);
  }

  // ---- epilogue: bias + tanh + dot with w_full, reduce over ATT ----
  // C layout: VGPR r, lanes 0-15: (M=r, N=lane); lanes 16-31: (M=8+r, N=lane-16)
  float part[2][8] = {};
#pragma unroll
  for (int t = 0; t < 4; ++t) {
    const int n = wave * 64 + t * 16 + m;
    const float bias = b_enc[n] + dec_out[b * ATT + n];
    const float wf   = w_full[n];
#pragma unroll
    for (int r = 0; r < 8; ++r) {
      part[0][r] += tanhf(acc[0][t][r] + bias) * wf;
      part[1][r] += tanhf(acc[1][t][r] + bias) * wf;
    }
  }

#pragma unroll
  for (int mt = 0; mt < 2; ++mt) {
#pragma unroll
    for (int r = 0; r < 8; ++r) {
      float v = part[mt][r];
      v += __shfl_xor(v, 1, 32);
      v += __shfl_xor(v, 2, 32);
      v += __shfl_xor(v, 4, 32);
      v += __shfl_xor(v, 8, 32);    // reduced over the 16 lanes of the half
      if (m == 0) atomicAdd(&s_logit[mt * 16 + r + hi * 8], v);  // ds_add_f32
    }
  }
  __syncthreads();

  if (tid < MROWS)
    logits[(size_t)b * NL + l0 + tid] = s_logit[tid] + b_full[0];
}

// ---------------------------------------------------------------------------
// Kernel 2: softmax over L, in place.  One block per batch row.
__global__ __launch_bounds__(256) void k_softmax(float* __restrict__ scores) {
  __shared__ float red[256];
  const int b = blockIdx.x, tid = threadIdx.x;
  float* row = scores + (size_t)b * NL;

  float x0 = row[tid], x1 = row[tid + 256], x2 = row[tid + 512], x3 = row[tid + 768];
  float mx = fmaxf(fmaxf(x0, x1), fmaxf(x2, x3));
  red[tid] = mx;
  __syncthreads();
  for (int s = 128; s > 0; s >>= 1) {
    if (tid < s) red[tid] = fmaxf(red[tid], red[tid + s]);
    __syncthreads();
  }
  mx = red[0];
  __syncthreads();

  float e0 = __expf(x0 - mx), e1 = __expf(x1 - mx);
  float e2 = __expf(x2 - mx), e3 = __expf(x3 - mx);
  red[tid] = e0 + e1 + e2 + e3;
  __syncthreads();
  for (int s = 128; s > 0; s >>= 1) {
    if (tid < s) red[tid] += red[tid + s];
    __syncthreads();
  }
  const float inv = 1.0f / red[0];
  row[tid]       = e0 * inv;
  row[tid + 256] = e1 * inv;
  row[tid + 512] = e2 * inv;
  row[tid + 768] = e3 * inv;
}

// ---------------------------------------------------------------------------
// Kernel 3a: partial weighted sums over L-chunks (memory-bound, coalesced).
// grid = (VIS/256, NB, 4).  part layout: [4][NB][VIS]
__global__ __launch_bounds__(256) void k_wsum_part(const float* __restrict__ vis,
                                                   const float* __restrict__ scores,
                                                   float* __restrict__ part) {
  __shared__ float s_sc[256];
  const int tid = threadIdx.x;
  const int vt  = blockIdx.x;   // v tile 0..3
  const int b   = blockIdx.y;
  const int lc  = blockIdx.z;   // l chunk 0..3

  s_sc[tid] = scores[(size_t)b * NL + lc * 256 + tid];
  __syncthreads();

  const int v = vt * 256 + tid;
  const float* base = vis + ((size_t)b * NL + lc * 256) * VIS + v;
  float acc = 0.0f;
#pragma unroll 4
  for (int l = 0; l < 256; ++l) acc += s_sc[l] * base[(size_t)l * VIS];
  part[((size_t)lc * NB + b) * VIS + v] = acc;
}

// Kernel 3b: deterministic reduce of the 4 partials -> att_output
__global__ __launch_bounds__(256) void k_wsum_reduce(const float* __restrict__ part,
                                                     float* __restrict__ out) {
  int idx = blockIdx.x * 256 + threadIdx.x;   // b*VIS + v over NB*VIS
  float s = part[idx]
          + part[(size_t)1 * NB * VIS + idx]
          + part[(size_t)2 * NB * VIS + idx]
          + part[(size_t)3 * NB * VIS + idx];
  out[idx] = s;
}

// ---------------------------------------------------------------------------
extern "C" void kernel_launch(void* const* d_in, const int* in_sizes, int n_in,
                              void* d_out, int out_size, void* d_ws, size_t ws_size,
                              hipStream_t stream) {
  const float* vis    = (const float*)d_in[0];  // [64,1024,1024]
  const float* dec_h  = (const float*)d_in[1];  // [64,512]
  const float* W_enc  = (const float*)d_in[2];  // [1024,512]
  const float* b_enc  = (const float*)d_in[3];  // [512]
  const float* W_dec  = (const float*)d_in[4];  // [512,512]
  const float* b_dec  = (const float*)d_in[5];  // [512]
  const float* w_full = (const float*)d_in[6];  // [512]
  const float* b_full = (const float*)d_in[7];  // [1]

  float* att_out    = (float*)d_out;                 // [64,1024]
  float* att_scores = (float*)d_out + NB * NL;       // [64,1024] (logits->softmax in place)

  // Workspace layout
  char*   ws      = (char*)d_ws;
  __bf16* Wt      = (__bf16*)ws;                              // 1 MB
  float*  dec_out = (float*)(ws + (size_t)(1u << 20));        // 128 KB
  float*  part    = (float*)(ws + (size_t)(1u << 20) + (128u << 10)); // 1 MB

  k_prep_wt<<<(ATT * VIS) / 256, 256, 0, stream>>>(W_enc, Wt);
  k_dec_proj<<<(NB * ATT) / 256, 256, 0, stream>>>(dec_h, W_dec, b_dec, dec_out);

  dim3 gGemm(NL / MROWS, NB);   // 32 x 64 blocks
  k_att_gemm<<<gGemm, 256, 0, stream>>>(vis, Wt, b_enc, dec_out, w_full, b_full,
                                        att_scores);

  k_softmax<<<NB, 256, 0, stream>>>(att_scores);

  dim3 gW(VIS / 256, NB, 4);
  k_wsum_part<<<gW, 256, 0, stream>>>(vis, att_scores, part);
  k_wsum_reduce<<<(NB * VIS) / 256, 256, 0, stream>>>(part, att_out);
}
